// RINModel_43954695307351
// MI455X (gfx1250) — matrix operand order repaired and checked by reading, not compiled, not run on hardware
//
#include <hip/hip_runtime.h>
#include <stdint.h>

// ---------------- problem constants ----------------
#define BATCH   8
#define SEQ     256
#define VOCAB   50257
#define LAY     2
#define NHID    256
#define DDIM    128
#define MROWS   (BATCH * SEQ)          // 2048
#define RES     4096
#define PHI_F   1.61803398874989485f
#define R2PI_F  651.8986469044033f     // RES / (2*pi)
#define STEP_F  0.00153398078788564f   // (2*pi) / RES
#define LOGITS_N ((size_t)BATCH * SEQ * VOCAB)

typedef __attribute__((ext_vector_type(16))) __bf16 v16bf;
typedef __attribute__((ext_vector_type(8)))  float  v8f;

// ---------------- helpers ----------------
__device__ __forceinline__ uint16_t f2bf(float x) {
    uint32_t u = __float_as_uint(x);
    u += 0x7FFFu + ((u >> 16) & 1u);          // round-to-nearest-even
    return (uint16_t)(u >> 16);
}

__device__ __forceinline__ int wrap_idx(float th) {
    int i = (int)rintf(th * R2PI_F) % RES;
    return (i < 0) ? i + RES : i;
}

// A fragment: 16x32 bf16 (MxK), row-major source, lda in elements.
// lane L: M = L%16, hi = L/16; VGPR j<4: K=2j+8hi ; j>=4: K=16+2(j-4)+8hi
__device__ __forceinline__ v16bf load_frag_a(const uint16_t* A, int lda,
                                             int m0, int kbase, int lane) {
    const int ml = lane & 15, hi = lane >> 4;
    const uint16_t* row = A + (size_t)(m0 + ml) * lda + kbase;
    union { uint32_t u[8]; v16bf v; } f;
#pragma unroll
    for (int j = 0; j < 8; ++j) {
        int kk = (j < 4) ? (2 * j + 8 * hi) : (16 + 2 * (j - 4) + 8 * hi);
        f.u[j] = *(const uint32_t*)(row + kk);
    }
    return f.v;
}

// B fragment: 32x16 bf16 (KxN) from B^T stored row-major (N rows, K contiguous).
// lane L: N = L%16, hi = L/16; VGPR j<4: K=2j+8hi ; j>=4: K=16+2(j-4)+8hi
__device__ __forceinline__ v16bf load_frag_b(const uint16_t* Bt, int ldb,
                                             int n0, int kbase, int lane,
                                             int nclamp) {
    int nr = n0 + (lane & 15);
    if (nr > nclamp) nr = nclamp;
    const int hi = lane >> 4;
    const uint16_t* row = Bt + (size_t)nr * ldb + kbase;
    union { uint32_t u[8]; v16bf v; } f;
#pragma unroll
    for (int j = 0; j < 8; ++j) f.u[j] = *(const uint32_t*)(row + ((j < 4) ? (2 * j + 8 * hi) : (16 + 2 * (j - 4) + 8 * hi)));
    return f.v;
}

// ---------------- kernels ----------------
__global__ void k_tables(float* sin_t, float* cos_t) {
    int i = blockIdx.x * blockDim.x + threadIdx.x;
    if (i < RES) {
        float a = (float)i * STEP_F;
        sin_t[i] = sinf(a);
        cos_t[i] = cosf(a);
    }
}

__global__ void k_cvt_bf16(const float* __restrict__ s, uint16_t* __restrict__ d, int n) {
    int i = blockIdx.x * blockDim.x + threadIdx.x;
    if (i < n) d[i] = f2bf(s[i]);
}

// elementwise recurrence: 1024 independent chains (b,d), 256 steps each
__global__ void k_scan(const int* __restrict__ ids, const float* __restrict__ emb,
                       const float* __restrict__ sin_t, const float* __restrict__ cos_t,
                       float* __restrict__ X, uint16_t* __restrict__ Xh,
                       float* __restrict__ hout /* = d_out + LOGITS_N */) {
    const int b = blockIdx.x;        // 0..7
    const int d = threadIdx.x;       // 0..127
    float hr = 0.f, hi = 0.f;
    for (int t = 0; t < SEQ; ++t) {
        const int id = ids[b * SEQ + t];
        const float w  = emb[(size_t)id * (2 * DDIM) + d];
        const float bc = emb[(size_t)id * (2 * DDIM) + DDIM + d];
        const float theta = (hr + hi) / (1.f + fabsf(w)) + bc + (float)t * PHI_F;
        const int idx = wrap_idx(theta);
        hi = sin_t[idx];
        hr = cos_t[idx];
        const float x = hr + hi;
        const size_t m = (size_t)b * SEQ + t;     // row ordering matches (B,S,V)
        X [m * DDIM + d] = x;
        Xh[m * DDIM + d] = f2bf(x);
    }
    hout[b * DDIM + d]                = hr;   // h_real
    hout[BATCH * DDIM + b * DDIM + d] = hi;   // h_imag
}

// stage A: TH = X @ W^T + bias + t*phi ; emit SN/CS (bf16).  M=2048,N=256,K=128
__global__ void k_th(const uint16_t* __restrict__ Xh, const uint16_t* __restrict__ Wb,
                     const float* __restrict__ bias,
                     const float* __restrict__ sin_t, const float* __restrict__ cos_t,
                     uint16_t* __restrict__ SNh, uint16_t* __restrict__ CSh) {
    const int lane = threadIdx.x & 31, wave = threadIdx.x >> 5;
    const int n0 = (blockIdx.x * 8 + wave) * 16;
    const int m0 = blockIdx.y * 16;
    v8f acc = {};
#pragma unroll
    for (int kc = 0; kc < DDIM; kc += 32) {
        v16bf a = load_frag_a(Xh, DDIM, m0, kc, lane);
        v16bf b = load_frag_b(Wb, DDIM, n0, kc, lane, NHID - 1);
        acc = __builtin_amdgcn_wmma_f32_16x16x32_bf16(false, a, false, b,
                                                      (short)0, acc, false, false);
    }
    const int hi = lane >> 4, n = n0 + (lane & 15);
    const float bn = bias[n];
#pragma unroll
    for (int r = 0; r < 8; ++r) {
        const int m = m0 + r + 8 * hi;
        const float th = acc[r] + bn + (float)(m & (SEQ - 1)) * PHI_F;
        const int idx = wrap_idx(th);
        SNh[(size_t)m * NHID + n] = f2bf(sin_t[idx]);
        CSh[(size_t)m * NHID + n] = f2bf(cos_t[idx]);
    }
}

// stage B: O = CS @ Wr^T + SN @ Wi^T ; X += silu(O).  M=2048,N=128,K=256
__global__ void k_o(const uint16_t* __restrict__ CSh, const uint16_t* __restrict__ SNh,
                    const uint16_t* __restrict__ Wrb, const uint16_t* __restrict__ Wib,
                    float* __restrict__ X, uint16_t* __restrict__ Xh) {
    const int lane = threadIdx.x & 31, wave = threadIdx.x >> 5;
    const int n0 = wave * 16;          // 8 waves cover N=128
    const int m0 = blockIdx.y * 16;
    v8f acc = {};
#pragma unroll
    for (int kc = 0; kc < NHID; kc += 32) {
        v16bf a = load_frag_a(CSh, NHID, m0, kc, lane);
        v16bf b = load_frag_b(Wrb, NHID, n0, kc, lane, DDIM - 1);
        acc = __builtin_amdgcn_wmma_f32_16x16x32_bf16(false, a, false, b,
                                                      (short)0, acc, false, false);
    }
#pragma unroll
    for (int kc = 0; kc < NHID; kc += 32) {
        v16bf a = load_frag_a(SNh, NHID, m0, kc, lane);
        v16bf b = load_frag_b(Wib, NHID, n0, kc, lane, DDIM - 1);
        acc = __builtin_amdgcn_wmma_f32_16x16x32_bf16(false, a, false, b,
                                                      (short)0, acc, false, false);
    }
    const int hi = lane >> 4, n = n0 + (lane & 15);
#pragma unroll
    for (int r = 0; r < 8; ++r) {
        const int m = m0 + r + 8 * hi;
        const float o = acc[r];
        const float x = X[(size_t)m * DDIM + n] + o / (1.f + expf(-o));
        X [(size_t)m * DDIM + n] = x;
        Xh[(size_t)m * DDIM + n] = f2bf(x);
    }
}

// logits = X @ out_w^T.  M=2048, N=50257, K=128.
// Each wave: 16(M) x 64(N) = 4 accumulators; A fragment loaded once per K-chunk
// and reused across 4 WMMAs (A traffic /4, 16 WMMAs per wave).
__global__ void k_logits(const uint16_t* __restrict__ Xh, const uint16_t* __restrict__ OWb,
                         float* __restrict__ out) {
    const int lane = threadIdx.x & 31, wave = threadIdx.x >> 5;
    const int nt_base = (blockIdx.x * 8 + wave) * 4;       // first of 4 N-tiles
    if (nt_base * 16 >= VOCAB) return;                     // wave-uniform exit
    const int m0 = blockIdx.y * 16;
    v8f acc[4] = {{}, {}, {}, {}};
#pragma unroll
    for (int kc = 0; kc < DDIM; kc += 32) {
        v16bf a = load_frag_a(Xh, DDIM, m0, kc, lane);
#pragma unroll
        for (int nt = 0; nt < 4; ++nt) {
            v16bf b = load_frag_b(OWb, DDIM, (nt_base + nt) * 16, kc, lane,
                                  VOCAB - 1);
            acc[nt] = __builtin_amdgcn_wmma_f32_16x16x32_bf16(false, a, false, b,
                                                              (short)0, acc[nt],
                                                              false, false);
        }
    }
    const int hi = lane >> 4, nl = lane & 15;
#pragma unroll
    for (int nt = 0; nt < 4; ++nt) {
        const int n = (nt_base + nt) * 16 + nl;
        if (n >= VOCAB) continue;
#pragma unroll
        for (int r = 0; r < 8; ++r) {
            const int m = m0 + r + 8 * hi;
            out[(size_t)m * VOCAB + n] = acc[nt][r];
        }
    }
}

// ---------------- launch ----------------
extern "C" void kernel_launch(void* const* d_in, const int* in_sizes, int n_in,
                              void* d_out, int out_size, void* d_ws, size_t ws_size,
                              hipStream_t stream) {
    const int*   ids   = (const int*)  d_in[0];
    const float* emb   = (const float*)d_in[1];
    const float* lW    = (const float*)d_in[2];   // (L, N, D)
    const float* lb    = (const float*)d_in[3];   // (L, N)
    const float* lWr   = (const float*)d_in[4];   // (L, D, N)
    const float* lWi   = (const float*)d_in[5];   // (L, D, N)
    const float* outw  = (const float*)d_in[6];   // (V, D)
    float* out = (float*)d_out;

    // workspace layout (byte offsets)
    char* ws = (char*)d_ws;
    float*    sin_t = (float*)(ws + 0);
    float*    cos_t = (float*)(ws + 16384);
    float*    X     = (float*)(ws + 32768);                       // 2048*128 f32
    uint16_t* Xh    = (uint16_t*)(ws + 32768 + 1048576);          // 2048*128 bf16
    uint16_t* CSh   = (uint16_t*)(ws + 1605632);                  // 2048*256 bf16
    uint16_t* SNh   = (uint16_t*)(ws + 2654208);                  // 2048*256 bf16
    uint16_t* Wb    = (uint16_t*)(ws + 3702784);                  // L*256*128 bf16
    uint16_t* Wrb   = (uint16_t*)(ws + 3833856);                  // L*128*256 bf16
    uint16_t* Wib   = (uint16_t*)(ws + 3964928);                  // L*128*256 bf16
    uint16_t* OWb   = (uint16_t*)(ws + 4096000);                  // V*128 bf16

    // 1) sin/cos tables
    k_tables<<<dim3((RES + 255) / 256), dim3(256), 0, stream>>>(sin_t, cos_t);

    // 2) weight conversion to bf16 (B^T layouts already k-contiguous in memory)
    {
        int nW = LAY * NHID * DDIM;                 // 65536
        k_cvt_bf16<<<dim3((nW + 255) / 256), dim3(256), 0, stream>>>(lW,  Wb,  nW);
        k_cvt_bf16<<<dim3((nW + 255) / 256), dim3(256), 0, stream>>>(lWr, Wrb, nW);
        k_cvt_bf16<<<dim3((nW + 255) / 256), dim3(256), 0, stream>>>(lWi, Wib, nW);
        int nO = VOCAB * DDIM;                      // 6,432,896
        k_cvt_bf16<<<dim3((nO + 255) / 256), dim3(256), 0, stream>>>(outw, OWb, nO);
    }

    // 3) elementwise recurrence -> X, Xh, final h
    k_scan<<<dim3(BATCH), dim3(DDIM), 0, stream>>>(ids, emb, sin_t, cos_t,
                                                   X, Xh, out + LOGITS_N);

    // 4) MLP layers (WMMA GEMMs, 8 waves per block, one 16x16 tile per wave)
    for (int i = 0; i < LAY; ++i) {
        k_th<<<dim3(NHID / 16 / 8, MROWS / 16), dim3(256), 0, stream>>>(
            Xh, Wb + (size_t)i * NHID * DDIM, lb + (size_t)i * NHID,
            sin_t, cos_t, SNh, CSh);
        k_o<<<dim3(1, MROWS / 16), dim3(256), 0, stream>>>(
            CSh, SNh, Wrb + (size_t)i * DDIM * NHID, Wib + (size_t)i * DDIM * NHID,
            X, Xh);
    }

    // 5) logits GEMM (write-bandwidth bound: 412 MB fp32 out)
    {
        int ntiles  = (VOCAB + 15) / 16;            // 3142
        int ngroups = (ntiles + 3) / 4;             // 786 wave-groups of 4 tiles
        k_logits<<<dim3((ngroups + 7) / 8, MROWS / 16), dim3(256), 0, stream>>>(
            Xh, OWb, out);
    }
}
